// DynamicTransformer_28028956573711
// MI455X (gfx1250) — matrix operand order repaired and checked by reading, not compiled, not run on hardware
//
#include <hip/hip_runtime.h>
#include <hip/hip_bf16.h>

// ---------------- types ----------------
typedef unsigned short u16;
typedef unsigned int   u32;
typedef __bf16 bf16;
typedef bf16  bf16x16 __attribute__((ext_vector_type(16)));
typedef float f32x8   __attribute__((ext_vector_type(8)));
typedef u16   u16x16  __attribute__((ext_vector_type(16)));
typedef u16   u16x4   __attribute__((ext_vector_type(4)));
typedef u32   u32x4   __attribute__((ext_vector_type(4)));
typedef int   i32x4   __attribute__((ext_vector_type(4)));
typedef int   i32x8   __attribute__((ext_vector_type(8)));

#define NB   16      // batch
#define CDIM 128     // dim
#define HID  256     // hidden
#define HW   16384   // 128*128
#define IMW  128
#define IMH  128

__device__ __forceinline__ u16 f2bf(float f) {
    // native f32 -> bf16 convert (RNE), avoids manual bit-twiddle VALU chains
    return __builtin_bit_cast(u16, (bf16)f);
}
__device__ __forceinline__ float bf2f(u16 h) {
    return __uint_as_float(((unsigned)h) << 16);
}

// ---------------- Tensor Data Mover: 2D bf16 tile -> LDS ----------------
// Loads a [128 rows x 32 elems] 2-byte tile (row stride = kd elements) into LDS,
// with TDM padding of 2 dwords after every 16 dwords (64B row) => LDS row
// stride 72B = 36 u16, matching the fragment-read layout below.
__device__ __forceinline__ void tdm_load_tile_a(const u16* gsrc, u16* lds_dst, int kd) {
    const u32 lds_off = (u32)(unsigned long long)(const void*)lds_dst;
    const unsigned long long ga = (unsigned long long)gsrc;
    u32x4 g0;
    g0.x = 1u;                                            // count=1 (valid descriptor)
    g0.y = lds_off;                                       // lds_addr
    g0.z = (u32)ga;                                       // global_addr[31:0]
    g0.w = (u32)((ga >> 32) & 0x01FFFFFFull) | (2u << 30);// global_addr[56:32] | type=2
    i32x8 g1;
    g1[0] = (int)((1u << 16)    // data_size = 1 -> 2 bytes
                | (1u << 20)    // pad_enable
                | (3u << 22)    // pad_interval = 16 dwords (one 64B row)
                | (1u << 25));  // pad_amount   = 2 dwords (8B)
    g1[1] = (int)((u32)kd << 16);   // tensor_dim0[15:0]  (row length in elems)
    g1[2] = (int)(128u << 16);      // tensor_dim1[15:0]  (rows)
    g1[3] = (int)(32u << 16);       // tile_dim0 = 32 elems
    g1[4] = 128;                    // tile_dim1 = 128 rows
    g1[5] = kd;                     // tensor_dim0_stride[31:0]
    g1[6] = 0;
    g1[7] = 0;
    const i32x4 z4 = (i32x4)0;
#if __has_include(<hip/amd_detail/amd_gfx1250_TDM.h>)
    const i32x8 z8 = (i32x8)0;
    __builtin_amdgcn_tensor_load_to_lds(g0, g1, z4, z4, z8, 0);
#else
    __builtin_amdgcn_tensor_load_to_lds(g0, g1, z4, z4, 0);
#endif
}

// ---------------- prep: weights f32 -> bf16, zero pooled ----------------
__global__ __launch_bounds__(256) void prep_kernel(
    const float* __restrict__ w_in, const float* __restrict__ w_out,
    u16* __restrict__ w_in_bf, u16* __restrict__ w_out_bf,
    float* __restrict__ pooled)
{
    int i = blockIdx.x * 256 + threadIdx.x;
    if (i < HID * CDIM) {
        w_in_bf[i] = f2bf(w_in[i]);
    } else if (i < 2 * HID * CDIM) {
        int j = i - HID * CDIM;
        w_out_bf[j] = f2bf(w_out[j]);
    } else if (i < 2 * HID * CDIM + NB * HID) {
        pooled[i - 2 * HID * CDIM] = 0.0f;
    }
}

// ---------------- WMMA tiled GEMM ----------------
// D[m][n] = sum_k A[m][k] * B[k][n] + bias[m], per batch plane of B.
//   A: bf16 row-major [MT][KD] (weights) -- staged via TDM tensor_load_to_lds
//   B: per-batch plane [KD][HW] (f32 source if BF32, else bf16) -- staged +
//      transposed to [n][k] by the workgroup (also does f32->bf16 convert)
// Block tile 128(M) x 128(N), BK=32, 8 waves in 4x2 layout,
// each wave owns 2x4 WMMA 16x16 subtiles.
template<int KD, int MT, bool BF32, bool POOL>
__global__ __launch_bounds__(256) void gemm_wmma(
    const u16*  __restrict__ Abf,
    const float* __restrict__ Bf32,
    const u16*  __restrict__ Bbf,
    const float* __restrict__ bias,
    u16*   __restrict__ out_bf,
    float* __restrict__ out_f32,
    float* __restrict__ pooled)
{
    constexpr int LDA = 36;   // u16 stride; matches TDM pad (64B row + 8B pad)
    constexpr int LDB = 36;   // u16 stride, padded against bank conflicts
    __shared__ u16 lA[128 * LDA];
    __shared__ u16 lB[128 * LDB];

    const int t    = threadIdx.x;
    const int lane = t & 31;
    const int wave = t >> 5;
    const int lo   = lane & 15;
    const int hi   = lane >> 4;
    const int wm   = wave >> 1;          // 0..3 -> 32-row slice of M
    const int wn   = wave & 1;           // 0..1 -> 64-col slice of N

    const int n_tile = blockIdx.x * 128;
    const int m_tile = blockIdx.y * 128;
    const int b      = blockIdx.z;
    const size_t bplane = (size_t)b * KD * HW;

    f32x8 acc[2][4];
#pragma unroll
    for (int mi = 0; mi < 2; ++mi)
#pragma unroll
        for (int ni = 0; ni < 4; ++ni)
            acc[mi][ni] = (f32x8)(0.0f);

    // cooperative B-load index mapping (constant across k loop)
    const int bk = t >> 3;               // 0..31   (K row)
    const int bn = (t & 7) * 16;         // 0..112  (N offset)

    for (int k0 = 0; k0 < KD; k0 += 32) {
        // ---- stage A tile [128][32] via Tensor Data Mover (wave 0 issues) ----
        if (wave == 0)
            tdm_load_tile_a(Abf + (size_t)m_tile * KD + k0, lA, KD);

        // ---- stage B tile transposed into [n][k] (all waves) ----
        {
            const size_t grow = bplane + (size_t)(k0 + bk) * HW + (n_tile + bn);
            if constexpr (BF32) {
#pragma unroll
                for (int j = 0; j < 16; ++j)
                    lB[(bn + j) * LDB + bk] = f2bf(Bf32[grow + j]);
            } else {
#pragma unroll
                for (int j = 0; j < 16; ++j)
                    lB[(bn + j) * LDB + bk] = Bbf[grow + j];
            }
            if (k0 + 32 < KD) {   // hint next K tile (global_prefetch_b8)
                const size_t nrow = bplane + (size_t)(k0 + 32 + bk) * HW + (n_tile + bn);
                if constexpr (BF32) __builtin_prefetch(Bf32 + nrow, 0, 1);
                else                __builtin_prefetch(Bbf + nrow, 0, 1);
            }
        }
        // TDM completion visible to whole WG: issuing wave drains TENSORcnt,
        // then everyone meets at the barrier.
        if (wave == 0)
            __builtin_amdgcn_s_wait_tensorcnt(0);
        __syncthreads();

        // ---- build fragments per ISA lane layouts ----
        bf16x16 afr[2], bfr[4];
#pragma unroll
        for (int mi = 0; mi < 2; ++mi) {
            const u16* p = &lA[(wm * 32 + mi * 16 + lo) * LDA];
            u16x16 v;
#pragma unroll
            for (int j = 0; j < 4; ++j) {
                // elems 0..7 -> K = 8*hi..8*hi+7 ; elems 8..15 -> K = 16+8*hi..
                const int off = (j < 2) ? (8 * hi + 4 * j) : (16 + 8 * hi + 4 * (j - 2));
                u16x4 c = *(const u16x4*)(p + off);
                v[4 * j + 0] = c[0]; v[4 * j + 1] = c[1];
                v[4 * j + 2] = c[2]; v[4 * j + 3] = c[3];
            }
            afr[mi] = __builtin_bit_cast(bf16x16, v);
        }
#pragma unroll
        for (int ni = 0; ni < 4; ++ni) {
            // B: elems 0..7 -> K = 8*hi..8*hi+7 ; elems 8..15 -> K = 16+8*hi.. (matches A)
            const u16* p = &lB[(wn * 64 + ni * 16 + lo) * LDB];
            u16x16 v;
#pragma unroll
            for (int j = 0; j < 4; ++j) {
                const int off = (j < 2) ? (8 * hi + 4 * j) : (16 + 8 * hi + 4 * (j - 2));
                u16x4 c = *(const u16x4*)(p + off);
                v[4 * j + 0] = c[0]; v[4 * j + 1] = c[1];
                v[4 * j + 2] = c[2]; v[4 * j + 3] = c[3];
            }
            bfr[ni] = __builtin_bit_cast(bf16x16, v);
        }

        // ---- 8 WMMAs ----
#pragma unroll
        for (int mi = 0; mi < 2; ++mi)
#pragma unroll
            for (int ni = 0; ni < 4; ++ni)
                acc[mi][ni] = __builtin_amdgcn_wmma_f32_16x16x32_bf16(
                    false, afr[mi], false, bfr[ni],
                    (short)0, acc[mi][ni], false, false);
        __syncthreads();
    }

    // ---- epilogue: bias (+ pooled butterfly) ----
#pragma unroll
    for (int mi = 0; mi < 2; ++mi) {
#pragma unroll
        for (int r = 0; r < 8; ++r) {
            const int m = m_tile + wm * 32 + mi * 16 + r + 8 * hi;  // D layout: M = r (+8 for hi lanes)
            const float bv = bias[m];
            if constexpr (POOL) {
                float psum = 0.0f;
#pragma unroll
                for (int ni = 0; ni < 4; ++ni) {
                    const float v = acc[mi][ni][r] + bv;
                    psum += v;
                    const size_t p = (size_t)n_tile + wn * 64 + ni * 16 + lo;
                    out_bf[((size_t)b * MT + m) * HW + p] = f2bf(v);
                }
                // reduce over the 16 N-lanes of this half-wave
                psum += __shfl_xor(psum, 1);
                psum += __shfl_xor(psum, 2);
                psum += __shfl_xor(psum, 4);
                psum += __shfl_xor(psum, 8);
                if (lo == 0) atomicAdd(&pooled[b * MT + m], psum);
            } else {
#pragma unroll
                for (int ni = 0; ni < 4; ++ni) {
                    const size_t p = (size_t)n_tile + wn * 64 + ni * 16 + lo;
                    out_f32[((size_t)b * MT + m) * HW + p] = acc[mi][ni][r] + bv;
                }
            }
        }
    }
}

// ---------------- kernel generator MLP (tiny) ----------------
__global__ __launch_bounds__(256) void kernelgen_kernel(
    const float* __restrict__ pooled,
    const float* __restrict__ w1, const float* __restrict__ b1,
    const float* __restrict__ w2, const float* __restrict__ b2,
    float* __restrict__ kout)
{
    __shared__ float sp[NB * HID];
    __shared__ float sh[NB * HID];
    const int t = threadIdx.x;
    const float inv = 1.0f / (float)HW;
#pragma unroll
    for (int j = 0; j < 16; ++j) {
        int i = t + j * 256;
        sp[i] = pooled[i] * inv;   // mean
    }
    __syncthreads();
    // h = relu(pooled @ w1^T + b1)
#pragma unroll 1
    for (int j = 0; j < 16; ++j) {
        int i = t + j * 256;
        int bb = i >> 8, o = i & 255;
        const float* prow = &sp[bb << 8];
        const float* wrow = &w1[(size_t)o << 8];
        float a = b1[o];
        for (int c = 0; c < HID; ++c) a = fmaf(prow[c], wrow[c], a);
        sh[i] = fmaxf(a, 0.0f);
    }
    __syncthreads();
    // kernels = h @ w2^T + b2  -> (B, HID*9)
#pragma unroll 1
    for (int j = 0; j < 144; ++j) {
        int i = t + j * 256;           // 36864 total
        int bb = i / (HID * 9);
        int q  = i - bb * (HID * 9);
        const float* hrow = &sh[bb << 8];
        const float* wrow = &w2[(size_t)q << 8];
        float a = b2[q];
        for (int c = 0; c < HID; ++c) a = fmaf(hrow[c], wrow[c], a);
        kout[i] = a;
    }
}

// ---------------- dynamic depthwise 3x3 + Mish ----------------
__global__ __launch_bounds__(256) void dwconv_mish_kernel(
    const u16* __restrict__ xh,       // bf16 (B,HID,128,128)
    const float* __restrict__ kerns,  // (B,HID,9)
    u16* __restrict__ yb)             // bf16 out
{
    __shared__ float tile[34 * IMW];
    const int t     = threadIdx.x;
    const int strip = blockIdx.x;          // 0..3  (32 rows each)
    const int c     = blockIdx.y;
    const int b     = blockIdx.z;
    const size_t plane = ((size_t)b * HID + c) * HW;
    const int r0 = strip * 32;

    // 3x3 kernel -> scalar (uniform) registers
    float kv[9];
    const float* kp = &kerns[((size_t)b * HID + c) * 9];
#pragma unroll
    for (int j = 0; j < 9; ++j) kv[j] = kp[j];

    // stage 34 x 128 haloed rows (zero-pad vertically)
#pragma unroll 1
    for (int j = 0; j < 17; ++j) {
        int e = t + j * 256;               // 4352 = 34*128
        int row = e >> 7, col = e & 127;
        int grow = r0 - 1 + row;
        float v = 0.0f;
        if (grow >= 0 && grow < IMH) v = bf2f(xh[plane + (size_t)grow * IMW + col]);
        tile[e] = v;
    }
    __syncthreads();

    const int col = t & 127;
#pragma unroll 1
    for (int j = 0; j < 16; ++j) {
        const int rl = 2 * j + (t >> 7);   // 0..31
        float a = 0.0f;
#pragma unroll
        for (int dy = 0; dy < 3; ++dy) {
#pragma unroll
            for (int dx = 0; dx < 3; ++dx) {
                const int cc = col + dx - 1;
                float v = (cc >= 0 && cc < IMW) ? tile[(rl + dy) * IMW + cc] : 0.0f;
                a = fmaf(v, kv[dy * 3 + dx], a);
            }
        }
        // Mish: a * tanh(softplus(a)); tanh(softplus(a)) = n/(n+2), n = e^a(e^a+2)
        const float e = __expf(fminf(a, 30.0f));
        const float n = e * (e + 2.0f);
        const float m = a * (n / (n + 2.0f));
        yb[plane + (size_t)(r0 + rl) * IMW + col] = f2bf(m);
    }
}

// ---------------- launch ----------------
extern "C" void kernel_launch(void* const* d_in, const int* in_sizes, int n_in,
                              void* d_out, int out_size, void* d_ws, size_t ws_size,
                              hipStream_t stream) {
    const float* x     = (const float*)d_in[0];
    const float* w_in  = (const float*)d_in[1];
    const float* b_in  = (const float*)d_in[2];
    const float* kg_w1 = (const float*)d_in[3];
    const float* kg_b1 = (const float*)d_in[4];
    const float* kg_w2 = (const float*)d_in[5];
    const float* kg_b2 = (const float*)d_in[6];
    const float* w_out = (const float*)d_in[7];
    const float* b_out = (const float*)d_in[8];
    float* out = (float*)d_out;

    // workspace layout
    char* ws = (char*)d_ws;
    const size_t XH_BYTES = (size_t)NB * HID * HW * sizeof(u16);  // 134,217,728
    u16*   xh      = (u16*)(ws);
    u16*   yb      = (u16*)(ws + XH_BYTES);
    float* pooled  = (float*)(ws + 2 * XH_BYTES);
    float* kerns   = (float*)(ws + 2 * XH_BYTES + (size_t)NB * HID * 4);
    u16*   w_in_bf = (u16*)(ws + 2 * XH_BYTES + (size_t)NB * HID * 4 + (size_t)NB * HID * 9 * 4);
    u16*   w_out_bf= w_in_bf + HID * CDIM;

    // 1. weight convert + pooled zero  (32768 + 32768 + 4096 = 69632)
    prep_kernel<<<dim3(272), dim3(256), 0, stream>>>(w_in, w_out, w_in_bf, w_out_bf, pooled);

    // 2. project_in GEMM (bf16 WMMA, f32 acc) + fused bias + global-pool sums
    gemm_wmma<CDIM, HID, true, true>
        <<<dim3(HW / 128, HID / 128, NB), dim3(256), 0, stream>>>(
            w_in_bf, x, nullptr, b_in, xh, nullptr, pooled);

    // 3. dynamic-kernel MLP
    kernelgen_kernel<<<dim3(1), dim3(256), 0, stream>>>(pooled, kg_w1, kg_b1, kg_w2, kg_b2, kerns);

    // 4. depthwise 3x3 dynamic conv + Mish
    dwconv_mish_kernel<<<dim3(4, HID, NB), dim3(256), 0, stream>>>(xh, kerns, yb);

    // 5. project_out GEMM (bf16 WMMA) + bias, f32 output
    gemm_wmma<HID, CDIM, false, false>
        <<<dim3(HW / 128, CDIM / 128, NB), dim3(256), 0, stream>>>(
            w_out_bf, nullptr, yb, b_out, nullptr, out, nullptr);
}
